// DiffAttnLayer_63745904607578
// MI455X (gfx1250) — compile-verified, matrix operand order checked by reading, not run on hardware
//
#include <hip/hip_runtime.h>
#include <math.h>

// ---------------------------------------------------------------------------
// MI455X / gfx1250 differential-attention transformer block.
// - all GEMM-class math via v_wmma_f32_16x16x32_bf16 (wave32 WMMA)
// - global->LDS tile movement via async-tensor path
//   (global_load_async_to_lds_b128, ASYNCcnt, s_wait_asynccnt)
// - GEMM: 128x128x64 tiles, double-buffered LDS, copy/math overlap
// ---------------------------------------------------------------------------

typedef __attribute__((ext_vector_type(16))) __bf16 v16bf;
typedef __attribute__((ext_vector_type(8)))  __bf16 v8bf;
typedef __attribute__((ext_vector_type(8)))  float  v8f;

#define WMMA_BF16 __builtin_amdgcn_wmma_f32_16x16x32_bf16

static __device__ __forceinline__ v8f v8f_zero() {
    v8f z = {0.f, 0.f, 0.f, 0.f, 0.f, 0.f, 0.f, 0.f};
    return z;
}

// Async global->LDS copy, 16B per lane. VDST = LDS byte address (low 32 bits
// of a __shared__ generic pointer), VADDR = 64-bit global address.
static __device__ __forceinline__ void async_load_b128(void* lds_ptr, const void* gptr) {
    uint32_t lds_off = (uint32_t)(uintptr_t)lds_ptr;
    asm volatile("global_load_async_to_lds_b128 %0, %1, off"
                 :: "v"(lds_off), "v"((uint64_t)(uintptr_t)gptr)
                 : "memory");
}
static __device__ __forceinline__ void wait_async0() {
    asm volatile("s_wait_asynccnt 0" ::: "memory");
}

// A-fragment (and contiguous B-fragment) load: row-major, 16B-aligned rows.
// Per-lane layout (16-bit A 16x32): halves 0..7 -> K=kb..kb+7, 8..15 -> K=kb+16..kb+23.
static __device__ __forceinline__ v16bf frag_row(const __bf16* rowp, int kbase) {
    v8bf lo = *(const v8bf*)(rowp + kbase);
    v8bf hi = *(const v8bf*)(rowp + kbase + 16);
    return __builtin_shufflevector(lo, hi, 0,1,2,3,4,5,6,7,8,9,10,11,12,13,14,15);
}

// B-fragment where K runs down a column of an LDS tile with compile-time stride.
template <int STRIDE>
static __device__ __forceinline__ v16bf frag_col(const __bf16* colp, int kbase) {
    v16bf v;
#pragma unroll
    for (int j = 0; j < 8; ++j) {
        v[j]     = colp[(kbase + j) * STRIDE];
        v[j + 8] = colp[(kbase + 16 + j) * STRIDE];
    }
    return v;
}

// ---------------------------------------------------------------------------
// fp32 -> bf16 conversion (weights)
// ---------------------------------------------------------------------------
__global__ void cvt_f32_bf16(const float* __restrict__ in, __bf16* __restrict__ out, int n) {
    int i = blockIdx.x * blockDim.x + threadIdx.x;
    if (i < n) out[i] = (__bf16)in[i];
}

// ---------------------------------------------------------------------------
// lambda = exp(lq1.lk1) - exp(lq2.lk2) + LAM_INIT   (HD = 64)
// ---------------------------------------------------------------------------
__global__ void lam_kernel(const float* lq1, const float* lk1,
                           const float* lq2, const float* lk2,
                           float lam_init, float* lam_out) {
    __shared__ float b1s[64], b2s[64];
    int t = threadIdx.x;
    b1s[t] = lq1[t] * lk1[t];
    b2s[t] = lq2[t] * lk2[t];
    __syncthreads();
    if (t == 0) {
        float d1 = 0.f, d2 = 0.f;
        for (int i = 0; i < 64; ++i) { d1 += b1s[i]; d2 += b2s[i]; }
        *lam_out = expf(d1) - expf(d2) + lam_init;
    }
}

// ---------------------------------------------------------------------------
// LayerNorm over 1024, fp32 in -> bf16 out. One block (256 thr) per token.
// ---------------------------------------------------------------------------
__global__ __launch_bounds__(256) void ln_bf16_kernel(
    const float* __restrict__ x, const float* __restrict__ w,
    const float* __restrict__ b, __bf16* __restrict__ out) {
    __shared__ float red[16];
    int row = blockIdx.x, tid = threadIdx.x;
    const float* xr = x + (size_t)row * 1024;
    float v0[4], s = 0.f, s2 = 0.f;
#pragma unroll
    for (int i = 0; i < 4; ++i) {
        float v = xr[tid + 256 * i];
        v0[i] = v; s += v; s2 += v * v;
    }
#pragma unroll
    for (int off = 16; off >= 1; off >>= 1) {
        s  += __shfl_xor(s,  off, 32);
        s2 += __shfl_xor(s2, off, 32);
    }
    int wave = tid >> 5;
    if ((tid & 31) == 0) { red[wave * 2] = s; red[wave * 2 + 1] = s2; }
    __syncthreads();
    float S = 0.f, S2 = 0.f;
#pragma unroll
    for (int i = 0; i < 8; ++i) { S += red[i * 2]; S2 += red[i * 2 + 1]; }
    float mean = S * (1.f / 1024.f);
    float var  = S2 * (1.f / 1024.f) - mean * mean;
    float rstd = rsqrtf(var + 1e-5f);
#pragma unroll
    for (int i = 0; i < 4; ++i) {
        int idx = tid + 256 * i;
        out[(size_t)row * 1024 + idx] = (__bf16)((v0[i] - mean) * rstd * w[idx] + b[idx]);
    }
}

// ---------------------------------------------------------------------------
// bf16 GEMM: C[M,N] = A[M,K] * B[K,N]. Block tile 128x128x64, 8 waves
// (2x4), wave tile 64x32 = 4x2 wmma tiles, 16 wmma per K-step.
// Double-buffered LDS, async global->LDS copies overlap WMMA.
// MODE epilogues:
//   0: out_bf16 = acc * scale
//   1: out_f32  = resid + acc (+ bias)
//   2: out_bf16 = gelu_exact(acc + bias)
// ---------------------------------------------------------------------------
template <int MODE>
__global__ __launch_bounds__(256) void gemm_bf16(
    const __bf16* __restrict__ A, const __bf16* __restrict__ Bw,
    const float* __restrict__ resid, const float* __restrict__ bias,
    void* __restrict__ outp, int M, int N, int K, float scale) {
    __shared__ __bf16 As[2][128][72];    // 64 + 8 pad, rows 144B (16B-aligned)
    __shared__ __bf16 Bs[2][64][136];    // 128 + 8 pad

    int tid = threadIdx.x;
    int lane = tid & 31, wave = tid >> 5;
    int wm = wave >> 2, wn = wave & 3;           // 2 x 4 wave grid
    int m0 = blockIdx.y * 128, n0 = blockIdx.x * 128;

    v8f acc[4][2];
#pragma unroll
    for (int mt = 0; mt < 4; ++mt)
#pragma unroll
        for (int nt = 0; nt < 2; ++nt) acc[mt][nt] = v8f_zero();

    int lrow = lane & 15;
    int kb = (lane >> 4) * 8;

    // issue async copies for one 128x64(A) + 64x128(B) tile
    auto issue_tile = [&](int buf, int k0) {
#pragma unroll
        for (int i = 0; i < 4; ++i) {            // A: 1024 x 16B chunks
            int c0 = tid * 4 + i;
            int r = c0 >> 3, cp = c0 & 7;
            async_load_b128(&As[buf][r][cp * 8],
                            A + (size_t)(m0 + r) * K + k0 + cp * 8);
        }
#pragma unroll
        for (int i = 0; i < 4; ++i) {            // B: 1024 x 16B chunks
            int c0 = tid * 4 + i;
            int r = c0 >> 4, cp = c0 & 15;
            async_load_b128(&Bs[buf][r][cp * 8],
                            Bw + (size_t)(k0 + r) * N + n0 + cp * 8);
        }
    };

    issue_tile(0, 0);
    wait_async0();
    __syncthreads();

    int nsteps = K >> 6;
    for (int t = 0; t < nsteps; ++t) {
        int cur = t & 1;
        if (t + 1 < nsteps) issue_tile(cur ^ 1, (t + 1) << 6);  // overlap with math

#pragma unroll
        for (int kk = 0; kk < 64; kk += 32) {
            v16bf af[4], bfv[2];
#pragma unroll
            for (int mt = 0; mt < 4; ++mt)
                af[mt] = frag_row(&As[cur][wm * 64 + mt * 16 + lrow][0], kk + kb);
#pragma unroll
            for (int nt = 0; nt < 2; ++nt)
                bfv[nt] = frag_col<136>(&Bs[cur][0][wn * 32 + nt * 16 + lrow], kk + kb);
#pragma unroll
            for (int mt = 0; mt < 4; ++mt)
#pragma unroll
                for (int nt = 0; nt < 2; ++nt)
                    acc[mt][nt] = WMMA_BF16(false, af[mt], false, bfv[nt],
                                            (short)0, acc[mt][nt], false, false);
        }
        wait_async0();
        __syncthreads();
    }

    int hi = lane >> 4;                          // C/D layout: lanes16-31 -> M+8
#pragma unroll
    for (int mt = 0; mt < 4; ++mt)
#pragma unroll
        for (int nt = 0; nt < 2; ++nt)
#pragma unroll
            for (int r = 0; r < 8; ++r) {
                int gm = m0 + wm * 64 + mt * 16 + r + 8 * hi;
                int gn = n0 + wn * 32 + nt * 16 + lrow;
                float v = acc[mt][nt][r];
                size_t idx = (size_t)gm * N + gn;
                if (MODE == 0) {
                    ((__bf16*)outp)[idx] = (__bf16)(v * scale);
                } else if (MODE == 1) {
                    v += resid[idx];
                    if (bias) v += bias[gn];
                    ((float*)outp)[idx] = v;
                } else {
                    float t2 = v + bias[gn];
                    float g = 0.5f * t2 * (1.0f + erff(t2 * 0.70710678118654752f));
                    ((__bf16*)outp)[idx] = (__bf16)g;
                }
            }
}

// ---------------------------------------------------------------------------
// Differential flash attention.
// grid = (N/64, H=8, B=2), block = 256 (8 waves).
// waves 0-3: sub-head 0 (rows 0..63), waves 4-7: sub-head 1; shared V tile.
// q/k/v tiles arrive via async global->LDS copies.
// ---------------------------------------------------------------------------
struct alignas(16) AttnSmem {
    union {
        struct {
            __bf16 qs[2][64][72];
            __bf16 ks[2][64][72];
            __bf16 vs[64][136];
            __bf16 ps[8][16][72];
        } a;
        float ost[2][64][132];
    };
};

__global__ __launch_bounds__(256) void diff_attn_kernel(
    const __bf16* __restrict__ q, const __bf16* __restrict__ k,
    const __bf16* __restrict__ v, const float* __restrict__ lam_p,
    const float* __restrict__ subln_w, __bf16* __restrict__ out,
    float one_minus_lam_init) {
    constexpr int N = 2048;
    __shared__ AttnSmem sm;

    int tid = threadIdx.x, lane = tid & 31, wave = tid >> 5;
    int hs = wave >> 2;          // sub-head select
    int rg = wave & 3;           // 16-row group within the 64-query tile
    int b = blockIdx.z, hh = blockIdx.y, q0 = blockIdx.x * 64;
    float lam = *lam_p;

    int lrow = lane & 15, hi = lane >> 4, kb = hi * 8;

    // async-load q (both sub-heads): 1024 x 16B chunks
#pragma unroll
    for (int i = 0; i < 4; ++i) {
        int c0 = tid * 4 + i;
        int h2 = c0 >> 9, r = (c0 >> 3) & 63, cp = c0 & 7;
        async_load_b128(&sm.a.qs[h2][r][cp * 8],
                        q + (((size_t)(b * N + q0 + r)) * 16 + (2 * hh + h2)) * 64 + cp * 8);
    }

    float mrow[8], lsum[8];
#pragma unroll
    for (int r = 0; r < 8; ++r) { mrow[r] = -1e30f; lsum[r] = 0.f; }
    v8f oacc[8];
#pragma unroll
    for (int vt = 0; vt < 8; ++vt) oacc[vt] = v8f_zero();

    for (int kt = 0; kt < N; kt += 64) {
        __syncthreads();                       // prior tile fully consumed
#pragma unroll
        for (int i = 0; i < 4; ++i) {          // k tile, both sub-heads
            int c0 = tid * 4 + i;
            int h2 = c0 >> 9, r = (c0 >> 3) & 63, cp = c0 & 7;
            async_load_b128(&sm.a.ks[h2][r][cp * 8],
                            k + (((size_t)(b * N + kt + r)) * 16 + (2 * hh + h2)) * 64 + cp * 8);
        }
#pragma unroll
        for (int i = 0; i < 4; ++i) {          // v tile 64x128
            int c0 = tid * 4 + i;
            int r = c0 >> 4, cp = c0 & 15;
            async_load_b128(&sm.a.vs[r][cp * 8],
                            v + (((size_t)(b * N + kt + r)) * 8 + hh) * 128 + cp * 8);
        }
        wait_async0();                         // covers q on first iteration too
        __syncthreads();

        // S strip (16 rows x 64 keys) = q * k^T
        v8f sacc[4];
#pragma unroll
        for (int nt = 0; nt < 4; ++nt) sacc[nt] = v8f_zero();
#pragma unroll
        for (int kk = 0; kk < 64; kk += 32) {
            v16bf af = frag_row(&sm.a.qs[hs][rg * 16 + lrow][0], kk + kb);
#pragma unroll
            for (int nt = 0; nt < 4; ++nt) {
                v16bf bfr = frag_row(&sm.a.ks[hs][nt * 16 + lrow][0], kk + kb);
                sacc[nt] = WMMA_BF16(false, af, false, bfr, (short)0, sacc[nt], false, false);
            }
        }

        // online softmax update
        float tmax[8];
#pragma unroll
        for (int r = 0; r < 8; ++r) {
            float m = sacc[0][r];
            m = fmaxf(m, sacc[1][r]); m = fmaxf(m, sacc[2][r]); m = fmaxf(m, sacc[3][r]);
            tmax[r] = m;
        }
#pragma unroll
        for (int off = 1; off < 16; off <<= 1)
#pragma unroll
            for (int r = 0; r < 8; ++r)
                tmax[r] = fmaxf(tmax[r], __shfl_xor(tmax[r], off, 32));

        float alpha[8], rsum[8];
#pragma unroll
        for (int r = 0; r < 8; ++r) {
            float mn = fmaxf(mrow[r], tmax[r]);
            alpha[r] = __expf(mrow[r] - mn);
            mrow[r] = mn; rsum[r] = 0.f;
        }
#pragma unroll
        for (int nt = 0; nt < 4; ++nt)
#pragma unroll
            for (int r = 0; r < 8; ++r) {
                float p = __expf(sacc[nt][r] - mrow[r]);
                rsum[r] += p;
                sm.a.ps[wave][r + 8 * hi][nt * 16 + lrow] = (__bf16)p;
            }
#pragma unroll
        for (int off = 1; off < 16; off <<= 1)
#pragma unroll
            for (int r = 0; r < 8; ++r)
                rsum[r] += __shfl_xor(rsum[r], off, 32);
#pragma unroll
        for (int r = 0; r < 8; ++r) lsum[r] = lsum[r] * alpha[r] + rsum[r];
#pragma unroll
        for (int vt = 0; vt < 8; ++vt)
#pragma unroll
            for (int r = 0; r < 8; ++r) oacc[vt][r] *= alpha[r];

        asm volatile("s_wait_dscnt 0" ::: "memory");   // wave-local P staging visible

        // O += P * V
#pragma unroll
        for (int kk = 0; kk < 64; kk += 32) {
            v16bf af = frag_row(&sm.a.ps[wave][lrow][0], kk + kb);
#pragma unroll
            for (int vt = 0; vt < 8; ++vt) {
                v16bf bfr = frag_col<136>(&sm.a.vs[0][vt * 16 + lrow], kk + kb);
                oacc[vt] = WMMA_BF16(false, af, false, bfr, (short)0, oacc[vt], false, false);
            }
        }
    }

    // stage normalized per-sub-head outputs, then combine + RMS-norm
    __syncthreads();
    float inv[8];
#pragma unroll
    for (int r = 0; r < 8; ++r) inv[r] = 1.0f / lsum[r];
#pragma unroll
    for (int vt = 0; vt < 8; ++vt)
#pragma unroll
        for (int r = 0; r < 8; ++r)
            sm.ost[hs][rg * 16 + r + 8 * hi][vt * 16 + lrow] = oacc[vt][r] * inv[r];
    __syncthreads();

    if (tid < 64) {
        float ss = 0.f;
        for (int d = 0; d < 128; ++d) {
            float val = sm.ost[0][tid][d] - lam * sm.ost[1][tid][d];
            ss += val * val;
        }
        float rr = rsqrtf(ss * (1.0f / 128.0f) + 1e-5f);
        size_t grow = (size_t)(b * N + q0 + tid);
        for (int d = 0; d < 128; ++d) {
            float val = sm.ost[0][tid][d] - lam * sm.ost[1][tid][d];
            out[grow * 1024 + hh * 128 + d] =
                (__bf16)(val * rr * subln_w[d] * one_minus_lam_init);
        }
    }
}

// ---------------------------------------------------------------------------
// Host orchestration
// ---------------------------------------------------------------------------
extern "C" void kernel_launch(void* const* d_in, const int* in_sizes, int n_in,
                              void* d_out, int out_size, void* d_ws, size_t ws_size,
                              hipStream_t stream) {
    constexpr int Bz = 2, N = 2048, EMB = 1024, FF = 4096;
    constexpr int T = Bz * N;                    // 4096 tokens
    (void)in_sizes; (void)n_in; (void)out_size; (void)ws_size;

    const float* x      = (const float*)d_in[0];
    const float* ln1_w  = (const float*)d_in[1];
    const float* ln1_b  = (const float*)d_in[2];
    const float* Wq     = (const float*)d_in[3];
    const float* Wk     = (const float*)d_in[4];
    const float* Wv     = (const float*)d_in[5];
    const float* Wo     = (const float*)d_in[6];
    const float* lq1    = (const float*)d_in[7];
    const float* lk1    = (const float*)d_in[8];
    const float* lq2    = (const float*)d_in[9];
    const float* lk2    = (const float*)d_in[10];
    const float* subln  = (const float*)d_in[11];
    const float* ln2_w  = (const float*)d_in[12];
    const float* ln2_b  = (const float*)d_in[13];
    const float* W1     = (const float*)d_in[14];
    const float* b1     = (const float*)d_in[15];
    const float* W2     = (const float*)d_in[16];
    const float* b2     = (const float*)d_in[17];

    float lam_init = 0.8f - 0.6f * expf(-0.3f);  // DEPTH = 1

    // workspace carve-out (~120 MB)
    char* ws = (char*)d_ws;
    size_t off = 0;
    auto take = [&](size_t bytes) -> char* {
        char* p = ws + off;
        off += (bytes + 255) & ~(size_t)255;
        return p;
    };
    __bf16* wq_b = (__bf16*)take((size_t)EMB * EMB * 2);
    __bf16* wk_b = (__bf16*)take((size_t)EMB * EMB * 2);
    __bf16* wv_b = (__bf16*)take((size_t)EMB * EMB * 2);
    __bf16* wo_b = (__bf16*)take((size_t)EMB * EMB * 2);
    __bf16* w1_b = (__bf16*)take((size_t)EMB * FF * 2);
    __bf16* w2_b = (__bf16*)take((size_t)FF * EMB * 2);
    __bf16* h1   = (__bf16*)take((size_t)T * EMB * 2);
    __bf16* qb   = (__bf16*)take((size_t)T * EMB * 2);
    __bf16* kbuf = (__bf16*)take((size_t)T * EMB * 2);
    __bf16* vbuf = (__bf16*)take((size_t)T * EMB * 2);
    __bf16* aout = (__bf16*)take((size_t)T * EMB * 2);
    __bf16* h2   = (__bf16*)take((size_t)T * EMB * 2);
    __bf16* ffn  = (__bf16*)take((size_t)T * FF * 2);
    float*  xmid = (float*)take((size_t)T * EMB * 4);
    float*  lamp = (float*)take(256);

    // scalar lambda + weight conversion
    lam_kernel<<<1, 64, 0, stream>>>(lq1, lk1, lq2, lk2, lam_init, lamp);
    int nw = EMB * EMB;
    cvt_f32_bf16<<<(nw + 255) / 256, 256, 0, stream>>>(Wq, wq_b, nw);
    cvt_f32_bf16<<<(nw + 255) / 256, 256, 0, stream>>>(Wk, wk_b, nw);
    cvt_f32_bf16<<<(nw + 255) / 256, 256, 0, stream>>>(Wv, wv_b, nw);
    cvt_f32_bf16<<<(nw + 255) / 256, 256, 0, stream>>>(Wo, wo_b, nw);
    int nff = EMB * FF;
    cvt_f32_bf16<<<(nff + 255) / 256, 256, 0, stream>>>(W1, w1_b, nff);
    cvt_f32_bf16<<<(nff + 255) / 256, 256, 0, stream>>>(W2, w2_b, nff);

    // LN1
    ln_bf16_kernel<<<T, 256, 0, stream>>>(x, ln1_w, ln1_b, h1);

    // QKV projections (q folds in HD^-0.5 = 1/8)
    dim3 gqkv(EMB / 128, T / 128);
    gemm_bf16<0><<<gqkv, 256, 0, stream>>>(h1, wq_b, nullptr, nullptr, qb,   T, EMB, EMB, 0.125f);
    gemm_bf16<0><<<gqkv, 256, 0, stream>>>(h1, wk_b, nullptr, nullptr, kbuf, T, EMB, EMB, 1.0f);
    gemm_bf16<0><<<gqkv, 256, 0, stream>>>(h1, wv_b, nullptr, nullptr, vbuf, T, EMB, EMB, 1.0f);

    // differential flash attention
    diff_attn_kernel<<<dim3(N / 64, 8, Bz), 256, 0, stream>>>(
        qb, kbuf, vbuf, lamp, subln, aout, 1.0f - lam_init);

    // output projection + residual (fp32)
    gemm_bf16<1><<<dim3(EMB / 128, T / 128), 256, 0, stream>>>(
        aout, wo_b, x, nullptr, xmid, T, EMB, EMB, 1.0f);

    // LN2
    ln_bf16_kernel<<<T, 256, 0, stream>>>(xmid, ln2_w, ln2_b, h2);

    // FFN: W1 + exact GELU, then W2 + b2 + residual -> d_out
    gemm_bf16<2><<<dim3(FF / 128, T / 128), 256, 0, stream>>>(
        h2, w1_b, nullptr, b1, ffn, T, FF, EMB, 1.0f);
    gemm_bf16<1><<<dim3(EMB / 128, T / 128), 256, 0, stream>>>(
        ffn, w2_b, xmid, b2, (float*)d_out, T, EMB, FF, 1.0f);
}
